// SPGG_21680994910874
// MI455X (gfx1250) — matrix operand order repaired and checked by reading, not compiled, not run on hardware
//
#include <hip/hip_runtime.h>

typedef __attribute__((ext_vector_type(16))) _Float16 v16h;
typedef __attribute__((ext_vector_type(8)))  float    v8f;

#define LGRID   2048
#define LMASK   2047
#define NCELLS  (LGRID * LGRID)

// ---------------------------------------------------------------------------
// Kernel A: nc = 5-point circular stencil of state (values 0..5 -> u8),
//           plus global sum of state (for the mean feature).
// ---------------------------------------------------------------------------
__global__ __launch_bounds__(256)
void spgg_nc_kernel(const int* __restrict__ state,
                    unsigned char* __restrict__ nc_out,
                    int* __restrict__ sum_out) {
    const int stride = gridDim.x * blockDim.x;
    int localSum = 0;
    for (int idx = blockIdx.x * blockDim.x + threadIdx.x; idx < NCELLS; idx += stride) {
        const int y = idx >> 11, x = idx & LMASK;
        const int xm = (x - 1) & LMASK, xp = (x + 1) & LMASK;
        const int ym = (y - 1) & LMASK, yp = (y + 1) & LMASK;
        const int s = state[idx];
        const int nc = s + state[(y << 11) | xm] + state[(y << 11) | xp]
                         + state[(ym << 11) | x] + state[(yp << 11) | x];
        nc_out[idx] = (unsigned char)nc;
        localSum += s;
    }
    // wave32 butterfly reduction, one atomic per wave
    #pragma unroll
    for (int off = 16; off > 0; off >>= 1) localSum += __shfl_xor(localSum, off, 32);
    if ((threadIdx.x & 31) == 0) atomicAdd(sum_out, localSum);
}

// ---------------------------------------------------------------------------
// Kernel B: fused reward + MLP. One wave == 16 row-contiguous cells.
//   h1 (16x64, f16) built directly in WMMA A-fragment layout,
//   h2 = relu(h1 @ W2 + b2) via 8x v_wmma_f32_16x16x32_f16,
//   heads (Wa: 64x2, Wc: 64x1) as per-lane dots + shfl_xor(16) reduce.
// ---------------------------------------------------------------------------
__global__ __launch_bounds__(256)
void spgg_mlp_kernel(const int* __restrict__ state,
                     const unsigned char* __restrict__ nc,
                     const int* __restrict__ sum_in,
                     const float* __restrict__ W1, const float* __restrict__ b1,
                     const float* __restrict__ W2, const float* __restrict__ b2,
                     const float* __restrict__ Wa, const float* __restrict__ ba,
                     const float* __restrict__ Wc, const float* __restrict__ bc,
                     float* __restrict__ out) {
    __shared__ float pre0[64];   // b1 + g * W1[2,:]
    __shared__ float w10s[64];   // W1[0,:]
    __shared__ float w11s[64];   // W1[1,:]
    __shared__ float b2s[64];
    __shared__ float WaS[128];   // row-major 64x2
    __shared__ float WcS[64];
    __shared__ float h2s[8][16 * 65];   // per-wave 16x64 tile, padded stride

    const int tid = threadIdx.x;
    const float g = (float)(*sum_in) * (1.0f / (float)NCELLS);   // global mean

    if (tid < 64) {
        pre0[tid] = b1[tid] + g * W1[2 * 64 + tid];
        w10s[tid] = W1[tid];
        w11s[tid] = W1[64 + tid];
        b2s[tid]  = b2[tid];
        WcS[tid]  = Wc[tid];
    }
    if (tid < 128) WaS[tid] = Wa[tid];
    __syncthreads();

    const int lane = tid & 31;
    const int wv   = tid >> 5;
    const int hi   = lane >> 4;      // lane half (ISA A/B fragment layout)
    const int l15  = lane & 15;
    float* h2w = &h2s[wv][0];

    const float ba0 = ba[0], ba1 = ba[1], bc0 = bc[0];

    // Hoisted W2 B-fragments. 16-bit B 32x16 layout: element e <-> K = 8*hi + e + (e & 8)
    // (same order as A); N = lane&15 within the 16-col block.
    v16h bf[4][2];
    #pragma unroll
    for (int nb = 0; nb < 4; ++nb)
        #pragma unroll
        for (int ks = 0; ks < 2; ++ks)
            #pragma unroll
            for (int e = 0; e < 16; ++e)
                bf[nb][ks][e] = (_Float16)W2[(ks * 32 + hi * 8 + e + (e & 8)) * 64 + nb * 16 + l15];

    const int nWaves = (gridDim.x * blockDim.x) >> 5;
    const int waveId = (blockIdx.x * blockDim.x + tid) >> 5;
    const int nTiles = NCELLS / 16;          // 128 tiles per grid row

    for (int t = waveId; t < nTiles; t += nWaves) {
        const int y  = t >> 7;
        const int x  = ((t & 127) << 4) + l15;   // cell column for row m = l15
        const int xm = (x - 1) & LMASK, xp = (x + 1) & LMASK;
        const int ym = (y - 1) & LMASK, yp = (y + 1) & LMASK;
        const int idx = (y << 11) | x;

        const int   s   = state[idx];
        const float sf  = (float)s;
        const float ncc = (float)nc[idx];
        const float n4  = (float)nc[(y << 11) | xm] + (float)nc[(y << 11) | xp]
                        + (float)nc[(ym << 11) | x] + (float)nc[(yp << 11) | x];
        // reward = neighbor_sum(nc) + nc - 6*state  (R=5 collapses the scale)
        const float reward = 2.0f * ncc + n4 - 6.0f * sf;

        // h1 = relu(feats @ W1 + b1), built in A-fragment layout:
        // element e <-> K = 8*hi + e + (e & 8); two K-splits (0..31, 32..63).
        v16h a0, a1;
        #pragma unroll
        for (int e = 0; e < 16; ++e) {
            const int kA = hi * 8 + e + (e & 8);
            float h0 = pre0[kA] + sf * w10s[kA] + ncc * w11s[kA];
            a0[e] = (_Float16)fmaxf(h0, 0.0f);
            const int kB = 32 + kA;
            float h1v = pre0[kB] + sf * w10s[kB] + ncc * w11s[kB];
            a1[e] = (_Float16)fmaxf(h1v, 0.0f);
        }

        // h2 accumulators: 4 x (16x16 f32), each fed by two K=32 WMMAs
        v8f cacc[4];
        #pragma unroll
        for (int nb = 0; nb < 4; ++nb) {
            v8f c = {};
            c = __builtin_amdgcn_wmma_f32_16x16x32_f16(false, a0, false, bf[nb][0],
                                                       (short)0, c, false, false);
            c = __builtin_amdgcn_wmma_f32_16x16x32_f16(false, a1, false, bf[nb][1],
                                                       (short)0, c, false, false);
            cacc[nb] = c;
        }

        // bias + relu, C-fragment (VGPR r -> row r + 8*hi) -> per-wave LDS tile
        #pragma unroll
        for (int nb = 0; nb < 4; ++nb) {
            const int n = nb * 16 + l15;
            const float bias = b2s[n];
            #pragma unroll
            for (int r = 0; r < 8; ++r)
                h2w[(r + 8 * hi) * 65 + n] = fmaxf(cacc[nb][r] + bias, 0.0f);
        }
        // wave-private LDS tile: only need DS-counter drain, no block barrier
        __builtin_amdgcn_wave_barrier();
        asm volatile("s_wait_dscnt 0" ::: "memory");
        __builtin_amdgcn_wave_barrier();

        // heads: lane (m = l15, half hi) covers 32 of 64 hidden units
        float p0 = 0.f, p1 = 0.f, pv = 0.f;
        const float* row = &h2w[l15 * 65 + hi * 32];
        #pragma unroll
        for (int j = 0; j < 32; ++j) {
            const int n = hi * 32 + j;
            const float h = row[j];
            p0 += h * WaS[2 * n];
            p1 += h * WaS[2 * n + 1];
            pv += h * WcS[n];
        }
        p0 += __shfl_xor(p0, 16, 32);
        p1 += __shfl_xor(p1, 16, 32);
        pv += __shfl_xor(pv, 16, 32);

        if (lane < 16) {
            const float l0 = p0 + ba0, l1 = p1 + ba1;
            const float mx = fmaxf(l0, l1);
            const float e0 = __expf(l0 - mx), e1 = __expf(l1 - mx);
            const float inv = 1.0f / (e0 + e1);
            out[2 * idx]              = e0 * inv;       // action_probs[:,0]
            out[2 * idx + 1]          = e1 * inv;       // action_probs[:,1]
            out[2 * NCELLS + idx]     = pv + bc0;       // state_value
            out[3 * NCELLS + idx]     = reward;         // reward
        }
    }
}

// ---------------------------------------------------------------------------
extern "C" void kernel_launch(void* const* d_in, const int* in_sizes, int n_in,
                              void* d_out, int out_size, void* d_ws, size_t ws_size,
                              hipStream_t stream) {
    const int*   state = (const int*)d_in[0];
    const float* W1    = (const float*)d_in[1];
    const float* b1    = (const float*)d_in[2];
    const float* W2    = (const float*)d_in[3];
    const float* b2    = (const float*)d_in[4];
    const float* Wa    = (const float*)d_in[5];
    const float* ba    = (const float*)d_in[6];
    const float* Wc    = (const float*)d_in[7];
    const float* bc    = (const float*)d_in[8];
    float* out = (float*)d_out;

    int*           sumPtr = (int*)d_ws;
    unsigned char* ncPtr  = (unsigned char*)d_ws + 64;

    hipMemsetAsync(d_ws, 0, 64, stream);   // zero the mean accumulator (capture-safe)
    spgg_nc_kernel<<<4096, 256, 0, stream>>>(state, ncPtr, sumPtr);
    spgg_mlp_kernel<<<4096, 256, 0, stream>>>(state, ncPtr, sumPtr,
                                              W1, b1, W2, b2, Wa, ba, Wc, bc, out);
    (void)in_sizes; (void)n_in; (void)out_size; (void)ws_size;
}
